// MambaLayer_41051297415323
// MI455X (gfx1250) — matrix operand order repaired and checked by reading, not compiled, not run on hardware
//
#include <hip/hip_runtime.h>
#include <hip/hip_fp16.h>

// ---------------------------------------------------------------------------
// Mamba layer (B=8, L=4096, d_model=128, d_inner=256, d_state=16) for MI455X
// gfx1250: wave32, WMMA f32<-f16 16x16x32 for all projections (compile-time
// shapes -> fully unrolled K loops, vectorized b128 fragment loads, packed
// f32->f16 converts), async global->LDS staging in the sequential scan.
// ---------------------------------------------------------------------------

typedef __attribute__((ext_vector_type(16))) _Float16 v16h;
typedef __attribute__((ext_vector_type(2)))  __fp16   h2v;   // cvt_pkrtz result type
typedef __attribute__((ext_vector_type(8)))  float    v8f;

#define B_SZ   8
#define L_SZ   4096
#define DMODEL 128
#define DINNER 256
#define DSTATE 16
#define DTRANK 8
#define NBC    40            // DT_RANK + 2*D_STATE
#define WIDTH  128
#define ROWS   (B_SZ * L_SZ) // 32768

union V16 { v16h v; h2v h2[8]; };   // same 32-byte layout, type-pun

// Convert 8 consecutive floats (16B-aligned) to 8 halves via packed converts.
// SCALE=true multiplies by m (0/1) to zero-pad K groups branchlessly.
template<bool SCALE>
__device__ __forceinline__ void cvt8(h2v* dst, const float* __restrict__ p, float m) {
  const float4 a = ((const float4*)p)[0];
  const float4 b = ((const float4*)p)[1];
  if constexpr (SCALE) {
    dst[0] = __builtin_amdgcn_cvt_pkrtz(a.x * m, a.y * m);
    dst[1] = __builtin_amdgcn_cvt_pkrtz(a.z * m, a.w * m);
    dst[2] = __builtin_amdgcn_cvt_pkrtz(b.x * m, b.y * m);
    dst[3] = __builtin_amdgcn_cvt_pkrtz(b.z * m, b.w * m);
  } else {
    dst[0] = __builtin_amdgcn_cvt_pkrtz(a.x, a.y);
    dst[1] = __builtin_amdgcn_cvt_pkrtz(a.z, a.w);
    dst[2] = __builtin_amdgcn_cvt_pkrtz(b.x, b.y);
    dst[3] = __builtin_amdgcn_cvt_pkrtz(b.z, b.w);
  }
}

// ===========================================================================
// WMMA GEMM:  C[M,N] = act( A[M,K'] (lda=LDA) * W[N,KREAL]^T + bias )
// K is the (padded) compute depth, KREAL the valid depth (KREAL<K only for
// dt_proj: K=32, KREAL=8). One wave computes a 16(M) x 64(N) strip.
// ACT: 0 = none (no bias), 1 = bias + softplus, 2 = bias + elu
// ===========================================================================
template<int ACT, int N, int K, int KREAL, int LDA, int LDC>
__global__ __launch_bounds__(128) void gemm_wmma(
    const float* __restrict__ A, const float* __restrict__ W,
    const float* __restrict__ bias, float* __restrict__ C, int M)
{
  constexpr int  NSTRIPS = (N + 63) / 64;
  constexpr int  NTILES  = (N >= 64) ? 4 : (N + 15) / 16;
  constexpr bool KPAD    = (KREAL != K);

  const int lane = threadIdx.x & 31;
  const int wave = threadIdx.x >> 5;
  const int r    = lane & 15;      // row (A) / col (B/C) index within tile
  const int hi   = lane >> 4;      // lane half selects K sub-range
  const int kb   = hi * 8;

  const int waveId = blockIdx.x * 4 + wave;
  const int m0  = (waveId / NSTRIPS) * 16;
  const int n0s = (waveId % NSTRIPS) * 64;
  if (m0 >= M) return;

  v8f acc[NTILES];
#pragma unroll
  for (int t = 0; t < NTILES; ++t) acc[t] = v8f{};

  const float* arow = A + (size_t)(m0 + r) * LDA;
  const float* wrow[NTILES];
#pragma unroll
  for (int t = 0; t < NTILES; ++t) {
    int wn = n0s + t * 16 + r;
    wn = (wn < N) ? wn : (N - 1);          // clamp: junk cols never stored
    wrow[t] = W + (size_t)wn * KREAL;
  }

#pragma unroll
  for (int k0 = 0; k0 < K; k0 += 32) {
    // A fragment (16-bit A 16x32, ISA 7.12.2):
    // lanes 0-15: K={k0..k0+7, k0+16..k0+23}; lanes 16-31: +8
    V16 af;
    cvt8<KPAD>(&af.h2[0], arow + k0 + kb,
               (!KPAD || (k0 + kb) < KREAL) ? 1.0f : 0.0f);
    cvt8<KPAD>(&af.h2[4], arow + k0 + 16 + kb,
               (!KPAD || (k0 + 16 + kb) < KREAL) ? 1.0f : 0.0f);
#pragma unroll
    for (int t = 0; t < NTILES; ++t) {
      // B fragment (32x16): lane = column; K={kw..kw+7, kw+16..kw+23}, kw=k0+8*half.
      const int kw = k0 + hi * 8;
      V16 bf;
      if constexpr (KPAD) {
        const bool v0 = kw < KREAL, v1 = (kw + 16) < KREAL;
        cvt8<true>(&bf.h2[0], v0 ? (wrow[t] + kw)     : W, v0 ? 1.0f : 0.0f);
        cvt8<true>(&bf.h2[4], v1 ? (wrow[t] + kw + 16) : W, v1 ? 1.0f : 0.0f);
      } else {
        cvt8<false>(&bf.h2[0], wrow[t] + kw,     1.0f);
        cvt8<false>(&bf.h2[4], wrow[t] + kw + 16, 1.0f);
      }
      acc[t] = __builtin_amdgcn_wmma_f32_16x16x32_f16(
          false, af.v, false, bf.v, (short)0, acc[t], false, false);
    }
  }

  // C/D layout: lane 0-15 -> N=lane, M=vgpr; lane 16-31 -> N=lane-16, M=vgpr+8
#pragma unroll
  for (int t = 0; t < NTILES; ++t) {
    const int n = n0s + t * 16 + r;
    if ((N % 16 == 0) || n < N) {
      const float bsv = (ACT != 0) ? bias[n] : 0.0f;
      float* cp = C + (size_t)(m0 + 8 * hi) * LDC + n;
#pragma unroll
      for (int v = 0; v < 8; ++v) {
        float val = acc[t][v] + bsv;
        if (ACT == 1)  // softplus
          val = (val > 20.0f) ? val : logf(1.0f + __expf(val));
        if (ACT == 2)  // elu
          val = (val > 0.0f) ? val : (__expf(val) - 1.0f);
        cp[(size_t)v * LDC] = val;
      }
    }
  }
}

template<int N> static inline int gemm_grid(int M) {
  return ((M / 16) * ((N + 63) / 64) + 3) / 4;   // 4 waves per block
}

// ===========================================================================
// Causal depthwise conv1d (width 4) + bias + SiLU over xm = xz[:, :, :256]
// ===========================================================================
__global__ __launch_bounds__(256) void conv_silu_kernel(
    const float* __restrict__ xz, const float* __restrict__ conv_w,
    const float* __restrict__ conv_b, float* __restrict__ xm)
{
  const int idx = blockIdx.x * 256 + threadIdx.x;  // over B*L*DINNER
  if (idx >= B_SZ * L_SZ * DINNER) return;
  const int d = idx & (DINNER - 1);
  const int l = (idx >> 8) & (L_SZ - 1);
  const int b = idx >> 20;
  const float* base = xz + (size_t)b * L_SZ * (2 * DINNER) + d;
  float acc = conv_b[d];
#pragma unroll
  for (int j = 0; j < 4; ++j) {
    const int li = l - 3 + j;
    if (li >= 0) acc += conv_w[d * 4 + j] * base[(size_t)li * (2 * DINNER)];
  }
  xm[idx] = acc / (1.0f + __expf(-acc));  // silu
}

// ===========================================================================
// Selective scan: sequential over L; one thread per (b, d) channel.
// h[16] and A-row in registers; B/C vectors staged into LDS via CDNA5
// async global->LDS copies (ASYNCcnt). Output overwrites xm in place.
// ===========================================================================
__global__ __launch_bounds__(64) void scan_kernel(
    const float* __restrict__ dt, const float* __restrict__ xdbc,
    const float* __restrict__ xz, const float* __restrict__ A_log,
    const float* __restrict__ Dp, float* __restrict__ xm)
{
  const int b = blockIdx.x;
  const int d = blockIdx.y * 64 + threadIdx.x;

  float A[DSTATE], h[DSTATE];
#pragma unroll
  for (int s = 0; s < DSTATE; ++s) {
    A[s] = -__expf(A_log[d * DSTATE + s]);
    h[s] = 0.0f;
  }
  const float Dd = Dp[d];

  const size_t rb = (size_t)b * L_SZ;
  const float* dt_p = dt + rb * DINNER + d;
  float*       x_p  = xm + rb * DINNER + d;
  const float* z_p  = xz + rb * (2 * DINNER) + DINNER + d;
  const float* bc_p = xdbc + rb * NBC;

  __shared__ float sBC[16 * 32];  // 16 timesteps x (16 B | 16 C)

  for (int lc = 0; lc < L_SZ; lc += 16) {
    const float* src = bc_p + (size_t)lc * NBC;
    __builtin_prefetch(src + 16 * NBC, 0, 0);                  // next B/C chunk
    __builtin_prefetch(dt_p + (size_t)(lc + 16) * DINNER, 0, 0);
    // Stage cols 8..39 of 16 rows ([B|C] is contiguous) via async LDS copies.
    for (int i = threadIdx.x; i < 16 * 32; i += 64) {
      const int t = i >> 5, c = i & 31;
      const unsigned ldsoff = (unsigned)(size_t)(&sBC[i]);     // LDS byte offset
      const float* g = src + t * NBC + DTRANK + c;
      asm volatile("global_load_async_to_lds_b32 %0, %1, off"
                   :: "v"(ldsoff), "v"(g) : "memory");
    }
    asm volatile("s_wait_asynccnt 0x0" ::: "memory");
    __syncthreads();
#pragma unroll 4
    for (int ll = 0; ll < 16; ++ll) {
      const int l = lc + ll;
      const float dtv = dt_p[(size_t)l * DINNER];   // already softplus'd
      const float xv  = x_p[(size_t)l * DINNER];
      const float dtx = dtv * xv;
      const float* bcl = sBC + ll * 32;
      float y = 0.0f;
#pragma unroll
      for (int s = 0; s < DSTATE; ++s) {
        h[s] = __expf(dtv * A[s]) * h[s] + dtx * bcl[s];
        y += h[s] * bcl[16 + s];
      }
      const float zv = z_p[(size_t)l * (2 * DINNER)];
      x_p[(size_t)l * DINNER] = (y + xv * Dd) * (zv / (1.0f + __expf(-zv)));
    }
    __syncthreads();
  }
}

// ===========================================================================
// Residual + LayerNorm + padding mask: one wave32 per 128-wide row.
// ===========================================================================
__global__ __launch_bounds__(256) void res_ln_mask_kernel(
    const float* __restrict__ xo, const float* __restrict__ h2,
    const float* __restrict__ ln_g, const float* __restrict__ ln_b,
    const unsigned char* __restrict__ mask, float* __restrict__ out)
{
  const int wave = threadIdx.x >> 5, lane = threadIdx.x & 31;
  const int row = blockIdx.x * 8 + wave;
  if (row >= ROWS) return;

  const float4 av = ((const float4*)(xo + (size_t)row * DMODEL))[lane];
  const float4 bv = ((const float4*)(h2 + (size_t)row * DMODEL))[lane];
  const float s0 = av.x + bv.x, s1 = av.y + bv.y;
  const float s2 = av.z + bv.z, s3 = av.w + bv.w;

  float sum = s0 + s1 + s2 + s3;
  float sq  = s0 * s0 + s1 * s1 + s2 * s2 + s3 * s3;
#pragma unroll
  for (int m = 16; m >= 1; m >>= 1) {   // wave32 tree reduction
    sum += __shfl_xor(sum, m, 32);
    sq  += __shfl_xor(sq,  m, 32);
  }
  const float mean = sum * (1.0f / DMODEL);
  const float var  = sq * (1.0f / DMODEL) - mean * mean;
  const float rstd = rsqrtf(var + 1e-5f);
  const float mk   = mask[row] ? 0.0f : 1.0f;  // True = padded -> zero

  const float4 g  = ((const float4*)ln_g)[lane];
  const float4 bb = ((const float4*)ln_b)[lane];
  float4 o;
  o.x = ((s0 - mean) * rstd * g.x + bb.x) * mk;
  o.y = ((s1 - mean) * rstd * g.y + bb.y) * mk;
  o.z = ((s2 - mean) * rstd * g.z + bb.z) * mk;
  o.w = ((s3 - mean) * rstd * g.w + bb.w) * mk;
  ((float4*)(out + (size_t)row * DMODEL))[lane] = o;
}

// ===========================================================================
// Host launch
// ===========================================================================
extern "C" void kernel_launch(void* const* d_in, const int* in_sizes, int n_in,
                              void* d_out, int out_size, void* d_ws, size_t ws_size,
                              hipStream_t stream) {
  (void)in_sizes; (void)n_in; (void)out_size; (void)ws_size;

  const float* x          = (const float*)d_in[0];
  const unsigned char* mk = (const unsigned char*)d_in[1]; // bool mask
  const float* in_proj_w  = (const float*)d_in[2];
  const float* conv_w     = (const float*)d_in[3];
  const float* conv_b     = (const float*)d_in[4];
  const float* x_proj_w   = (const float*)d_in[5];
  const float* dt_proj_w  = (const float*)d_in[6];
  const float* dt_proj_b  = (const float*)d_in[7];
  const float* A_log      = (const float*)d_in[8];
  const float* Dp         = (const float*)d_in[9];
  const float* out_proj_w = (const float*)d_in[10];
  const float* ln_g       = (const float*)d_in[11];
  const float* ln_b       = (const float*)d_in[12];
  const float* w1         = (const float*)d_in[13];
  const float* b1         = (const float*)d_in[14];
  const float* w2         = (const float*)d_in[15];
  const float* b2         = (const float*)d_in[16];

  // Workspace layout (floats), total = ROWS*(512+256+40+256+128) ~ 156 MB.
  float* ws   = (float*)d_ws;
  float* xz   = ws;                          // [ROWS,512]  xm | z
  float* xm   = xz   + (size_t)ROWS * 512;   // [ROWS,256]  conv+silu; scan out in place
  float* xdbc = xm   + (size_t)ROWS * 256;   // [ROWS,40]   dt_in | B | C
  float* dt   = xdbc + (size_t)ROWS * 40;    // [ROWS,256]  softplus(dt); reused for h1/h2
  float* xo   = dt   + (size_t)ROWS * 256;   // [ROWS,128]  mamba output
  float* h1   = dt;                          // [ROWS,128]  (dt dead after scan)
  float* h2   = dt   + (size_t)ROWS * 128;   // [ROWS,128]

  // 1) in_proj: xz = x @ in_proj_w^T            [32768,128] x [512,128]^T
  gemm_wmma<0, 512, 128, 128, 128, 512>
      <<<gemm_grid<512>(ROWS), 128, 0, stream>>>(x, in_proj_w, nullptr, xz, ROWS);

  // 2) causal depthwise conv + bias + silu
  conv_silu_kernel<<<(ROWS * DINNER) / 256, 256, 0, stream>>>(xz, conv_w, conv_b, xm);

  // 3) x_proj: xdbc = xm @ x_proj_w^T           [32768,256] x [40,256]^T
  gemm_wmma<0, NBC, 256, 256, 256, NBC>
      <<<gemm_grid<NBC>(ROWS), 128, 0, stream>>>(xm, x_proj_w, nullptr, xdbc, ROWS);

  // 4) dt_proj + softplus (K padded 8 -> 32)
  gemm_wmma<1, DINNER, 32, DTRANK, NBC, DINNER>
      <<<gemm_grid<DINNER>(ROWS), 128, 0, stream>>>(xdbc, dt_proj_w, dt_proj_b, dt, ROWS);

  // 5) selective scan (sequential over L), writes (y + x*D)*silu(z) into xm
  {
    dim3 grid(B_SZ, DINNER / 64);
    scan_kernel<<<grid, 64, 0, stream>>>(dt, xdbc, xz, A_log, Dp, xm);
  }

  // 6) out_proj: xo = xm @ out_proj_w^T         [32768,256] x [128,256]^T
  gemm_wmma<0, DMODEL, 256, 256, 256, DMODEL>
      <<<gemm_grid<DMODEL>(ROWS), 128, 0, stream>>>(xm, out_proj_w, nullptr, xo, ROWS);

  // 7) FFN: h1 = elu(xo@w1^T + b1); h2 = elu(h1@w2^T + b2)
  gemm_wmma<2, WIDTH, 128, 128, DMODEL, WIDTH>
      <<<gemm_grid<WIDTH>(ROWS), 128, 0, stream>>>(xo, w1, b1, h1, ROWS);
  gemm_wmma<2, DMODEL, 128, 128, WIDTH, DMODEL>
      <<<gemm_grid<DMODEL>(ROWS), 128, 0, stream>>>(h1, w2, b2, h2, ROWS);

  // 8) residual + layernorm + mask -> d_out
  res_ln_mask_kernel<<<ROWS / 8, 256, 0, stream>>>(xo, h2, ln_g, ln_b, mk, (float*)d_out);
}
